// PointerGeneratorNetwork_87574383165644
// MI455X (gfx1250) — matrix operand order repaired and checked by reading, not compiled, not run on hardware
//
#include <hip/hip_runtime.h>
#include <hip/hip_bf16.h>
#include <cstdint>

// Problem dims (B, T, S, D) = (16, 512, 2048, 768)
#define Bc 16
#define Tc 512
#define Sc 2048
#define Dc 768

typedef __attribute__((ext_vector_type(16))) _Float16 v16h;
typedef __attribute__((ext_vector_type(8)))  _Float16 v8h;
typedef __attribute__((ext_vector_type(4)))  _Float16 v4h;
typedef __attribute__((ext_vector_type(8)))  float    v8f;

// ---------------------------------------------------------------------------
// CDNA5 async global->LDS copy (ASYNCcnt path, GV addressing mode).
// Each active lane moves 16 bytes: LDS[lds_addr(lane)] = MEM[gaddr(lane)].
// ---------------------------------------------------------------------------
__device__ __forceinline__ void async_copy_b128(uint32_t lds_byte_addr, const void* gaddr) {
    asm volatile("global_load_async_to_lds_b128 %0, %1, off"
                 :: "v"(lds_byte_addr), "v"(gaddr) : "memory");
}
__device__ __forceinline__ void wait_async_zero() {
    asm volatile("s_wait_asynccnt 0x0" ::: "memory");
}
__device__ __forceinline__ uint32_t lds_addr_of(const void* p) {
    // LDS aperture: low 32 bits of a generic shared-pointer == LDS byte offset.
    return (uint32_t)(uintptr_t)p;
}

// ---------------------------------------------------------------------------
// K1: fused GEMVs. enc_score[b,s] = E[b,s,:]·w_e ; dec_score[b,t] = H[b,t,:]·w_d
// ---------------------------------------------------------------------------
__global__ __launch_bounds__(256) void k1_scores(const float* __restrict__ dec,
                                                 const float* __restrict__ enc,
                                                 const float* __restrict__ w_ptr,
                                                 float* __restrict__ enc_score,
                                                 float* __restrict__ dec_score) {
    const int gwave = (int)((blockIdx.x * blockDim.x + threadIdx.x) >> 5);
    const int lane  = (int)(threadIdx.x & 31);
    const int NENC  = Bc * Sc;

    const float4* src4;
    const float4* w4;
    float* out;
    if (gwave < NENC) {
        src4 = (const float4*)(enc + (size_t)gwave * Dc);
        w4   = (const float4*)(w_ptr + Dc);
        out  = enc_score + gwave;
    } else {
        int r = gwave - NENC;
        if (r >= Bc * Tc) return;
        src4 = (const float4*)(dec + (size_t)r * Dc);
        w4   = (const float4*)w_ptr;
        out  = dec_score + r;
    }

    float acc = 0.f;
#pragma unroll
    for (int i = 0; i < Dc / 4 / 32; ++i) {
        int idx = i * 32 + lane;
        float4 a = src4[idx], w = w4[idx];
        acc += a.x * w.x + a.y * w.y + a.z * w.z + a.w * w.w;
    }
#pragma unroll
    for (int off = 16; off >= 1; off >>= 1)
        acc += __shfl_down(acc, off, 32);
    if (lane == 0) *out = acc;
}

// ---------------------------------------------------------------------------
// K2: per-(b,t) masked softmax over S. Writes f32 pointer_weights (output) and
// an f16 shadow copy (GEMM A operand) in the same pass.
// ---------------------------------------------------------------------------
__global__ __launch_bounds__(256) void k2_softmax(const float* __restrict__ enc_score,
                                                  const float* __restrict__ dec_score,
                                                  const int*   __restrict__ mask,
                                                  const float* __restrict__ b_ptr,
                                                  float* __restrict__ pw,
                                                  _Float16* __restrict__ pw16) {
    const int b = (int)blockIdx.x / Tc;
    const int t = (int)blockIdx.x % Tc;
    const int tid = (int)threadIdx.x;

    const float ds   = dec_score[b * Tc + t] + b_ptr[0];
    const float* es  = enc_score + (size_t)b * Sc;
    const int*   msk = mask      + (size_t)b * Sc;

    float v[8];
    float mx = -3.0e38f;
#pragma unroll
    for (int i = 0; i < 8; ++i) {
        int s = tid + i * 256;
        float val = (msk[s] != 0) ? (es[s] + ds) : -1.0e9f;
        v[i] = val;
        mx = fmaxf(mx, val);
    }

    __shared__ float red[256];
    red[tid] = mx;
    __syncthreads();
#pragma unroll
    for (int off = 128; off >= 1; off >>= 1) {
        if (tid < off) red[tid] = fmaxf(red[tid], red[tid + off]);
        __syncthreads();
    }
    const float rowmax = red[0];
    __syncthreads();

    float sum = 0.f;
#pragma unroll
    for (int i = 0; i < 8; ++i) {
        v[i] = __expf(v[i] - rowmax);
        sum += v[i];
    }
    red[tid] = sum;
    __syncthreads();
#pragma unroll
    for (int off = 128; off >= 1; off >>= 1) {
        if (tid < off) red[tid] += red[tid + off];
        __syncthreads();
    }
    const float inv = 1.0f / red[0];

    float*    out   = pw   + ((size_t)b * Tc + t) * Sc;
    _Float16* out16 = pw16 + ((size_t)b * Tc + t) * Sc;
#pragma unroll
    for (int i = 0; i < 8; ++i) {
        float w = v[i] * inv;
        out[tid + i * 256]   = w;
        out16[tid + i * 256] = (_Float16)w;
    }
}

// ---------------------------------------------------------------------------
// K2b: convert + transpose encoder_outputs: E16T[b][d][s] = f16(E[b][s][d]).
// 64x64 LDS tile transpose, coalesced on both global sides.
// Makes the GEMM B operand f16 with K(=s) contiguous.
// ---------------------------------------------------------------------------
__global__ __launch_bounds__(256) void k2b_convT(const float* __restrict__ enc,
                                                 _Float16* __restrict__ e16t) {
    __shared__ _Float16 tile[64][66];           // 66-half row pad: conflict-free
    const int tid   = (int)threadIdx.x;
    const int tilesD = Dc / 64;                  // 12
    const int tilesS = Sc / 64;                  // 32
    int blk = (int)blockIdx.x;
    int b   = blk / (tilesD * tilesS);
    int r   = blk % (tilesD * tilesS);
    int s0  = (r / tilesD) * 64;
    int d0  = (r % tilesD) * 64;

    const float* src = enc + ((size_t)b * Sc + s0) * Dc + d0;
#pragma unroll
    for (int j = 0; j < 4; ++j) {
        int row = (tid >> 4) + j * 16;           // s-local
        int c4  = (tid & 15) * 4;                // d-local
        float4 v = *(const float4*)(src + (size_t)row * Dc + c4);
        tile[row][c4 + 0] = (_Float16)v.x;
        tile[row][c4 + 1] = (_Float16)v.y;
        tile[row][c4 + 2] = (_Float16)v.z;
        tile[row][c4 + 3] = (_Float16)v.w;
    }
    __syncthreads();

    _Float16* dst = e16t + ((size_t)b * Dc + d0) * Sc + s0;
#pragma unroll
    for (int j = 0; j < 4; ++j) {
        int dl = (tid >> 4) + j * 16;            // d-local (output row)
        int sl = (tid & 15) * 4;                 // s-local
        v4h o;
        o[0] = tile[sl + 0][dl];
        o[1] = tile[sl + 1][dl];
        o[2] = tile[sl + 2][dl];
        o[3] = tile[sl + 3][dl];
        *(v4h*)(dst + (size_t)dl * Sc + sl) = o; // 8B coalesced store
    }
}

// ---------------------------------------------------------------------------
// K3: context[b] = PW[b] (512x2048) @ E[b] (2048x768), f16 WMMA, f32 accum.
// Both operands pre-converted f16 with K contiguous (PW16 row-major [M][K],
// E16T row-major [N][K]; row stride = Sc halves for both).
// Block tile 128x128, 8 waves (2M x 4N), wave tile 64x32 = 4x2 WMMA frags.
// Double-buffered LDS staged via global_load_async_to_lds_b128: tile i+1 DMAs
// while tile i computes; drained with s_wait_asynccnt 0 + barrier.
// LDS row stride 40 halves (80B): 16B-aligned rows, conflict-free b128 reads.
// ---------------------------------------------------------------------------
#define LSTR 40
__global__ __launch_bounds__(256) void k3_ctx_gemm(const _Float16* __restrict__ pw16,
                                                   const _Float16* __restrict__ e16t,
                                                   float* __restrict__ ctx) {
    __shared__ __align__(16) _Float16 Atile[2][128 * LSTR];   // 2 x 10 KB
    __shared__ __align__(16) _Float16 Btile[2][128 * LSTR];   // 2 x 10 KB

    const int blk   = (int)blockIdx.x;
    const int b     = blk / 24;
    const int rem   = blk % 24;
    const int Mbase = (rem / 6) * 128;
    const int Nbase = (rem % 6) * 128;

    const _Float16* Ag = pw16 + ((size_t)b * Tc + Mbase) * Sc;  // [M][K], stride Sc
    const _Float16* Bg = e16t + ((size_t)b * Dc + Nbase) * Sc;  // [N][K], stride Sc

    const int tid   = (int)threadIdx.x;
    const int wid   = tid >> 5;
    const int lane  = tid & 31;
    const int waveM = wid & 1;
    const int waveN = wid >> 1;
    const int half  = lane >> 4;
    const int l16   = lane & 15;

    // stage one 128x32-half tile (8 KB) from global (f16) into LDS: 512 16B chunks
    auto stage = [&](const _Float16* g, _Float16* lds, int kt) {
#pragma unroll
        for (int j = 0; j < 2; ++j) {
            int c    = tid + j * 256;
            int row  = c >> 2;                 // 4 chunks per 32-half row
            int col8 = (c & 3) * 8;            // halves
            async_copy_b128(lds_addr_of(lds + row * LSTR + col8),
                            g + (size_t)row * Sc + kt + col8);
        }
    };

    v8f acc[4][2];
#pragma unroll
    for (int mt = 0; mt < 4; ++mt)
#pragma unroll
        for (int nt = 0; nt < 2; ++nt)
#pragma unroll
            for (int i = 0; i < 8; ++i) acc[mt][nt][i] = 0.f;

    // prologue: tile 0 into buffer 0
    stage(Ag, Atile[0], 0);
    stage(Bg, Btile[0], 0);
    wait_async_zero();
    __syncthreads();

    const int NT = Sc / 32;                    // 64 K-chunks
    const int kloA = half * 16;                // A frag: halves klo..klo+15
    const int kloB = half * 16;                // B frag: halves klo..klo+15

    for (int i = 0; i < NT; ++i) {
        const int cur = i & 1;
        if (i + 1 < NT) {                      // overlap next tile DMA with compute
            stage(Ag, Atile[1 - cur], (i + 1) * 32);
            stage(Bg, Btile[1 - cur], (i + 1) * 32);
        }

        // fragments: pure ds_load_b128 pairs, no VALU converts
        v16h afrag[4];
#pragma unroll
        for (int mt = 0; mt < 4; ++mt) {
            const _Float16* ap = &Atile[cur][(waveM * 64 + mt * 16 + l16) * LSTR];
            v8h lo = *(const v8h*)(ap + kloA);
            v8h hi = *(const v8h*)(ap + kloA + 8);
            v16h a;
#pragma unroll
            for (int q = 0; q < 8; ++q) { a[q] = lo[q]; a[8 + q] = hi[q]; }
            afrag[mt] = a;
        }
        v16h bfrag[2];
#pragma unroll
        for (int nt = 0; nt < 2; ++nt) {
            const _Float16* bp = &Btile[cur][(waveN * 32 + nt * 16 + l16) * LSTR];
            v8h lo = *(const v8h*)(bp + kloB);
            v8h hi = *(const v8h*)(bp + kloB + 8);
            v16h bb;
#pragma unroll
            for (int q = 0; q < 8; ++q) { bb[q] = lo[q]; bb[8 + q] = hi[q]; }
            bfrag[nt] = bb;
        }

#pragma unroll
        for (int mt = 0; mt < 4; ++mt)
#pragma unroll
            for (int nt = 0; nt < 2; ++nt)
                acc[mt][nt] = __builtin_amdgcn_wmma_f32_16x16x32_f16(
                    false, afrag[mt], false, bfrag[nt],
                    (short)0, acc[mt][nt], false, false);

        if (i + 1 < NT) {
            wait_async_zero();                 // tile i+1 landed (DMA'd during compute)
            __syncthreads();                   // all waves done + buffers consistent
        }
    }

    // store C: 16x16 f32 layout -> VGPR r: M = r + 8*half, N = l16
#pragma unroll
    for (int mt = 0; mt < 4; ++mt) {
        int rowb = Mbase + waveM * 64 + mt * 16 + half * 8;
#pragma unroll
        for (int nt = 0; nt < 2; ++nt) {
            int col = Nbase + waveN * 32 + nt * 16 + l16;
            float* cp = ctx + ((size_t)b * Tc + rowb) * Dc + col;
#pragma unroll
            for (int r = 0; r < 8; ++r)
                cp[(size_t)r * Dc] = acc[mt][nt][r];
        }
    }
}

// ---------------------------------------------------------------------------
// K4: p_gen[b,t] = sigmoid(H·g_d + context·g_e + b_gen). One wave32 per row.
// ---------------------------------------------------------------------------
__global__ __launch_bounds__(256) void k4_pgen(const float* __restrict__ dec,
                                               const float* __restrict__ ctx,
                                               const float* __restrict__ w_gen,
                                               const float* __restrict__ b_gen,
                                               float* __restrict__ pgen) {
    const int row  = (int)((blockIdx.x * blockDim.x + threadIdx.x) >> 5);
    const int lane = (int)(threadIdx.x & 31);
    if (row >= Bc * Tc) return;

    const float4* dp = (const float4*)(dec + (size_t)row * Dc);
    const float4* cp = (const float4*)(ctx + (size_t)row * Dc);
    const float4* gd = (const float4*)w_gen;
    const float4* ge = (const float4*)(w_gen + Dc);

    float acc = 0.f;
#pragma unroll
    for (int i = 0; i < Dc / 4 / 32; ++i) {
        int idx = i * 32 + lane;
        float4 a = dp[idx], wa = gd[idx];
        acc += a.x * wa.x + a.y * wa.y + a.z * wa.z + a.w * wa.w;
        float4 c = cp[idx], wc = ge[idx];
        acc += c.x * wc.x + c.y * wc.y + c.z * wc.z + c.w * wc.w;
    }
#pragma unroll
    for (int off = 16; off >= 1; off >>= 1)
        acc += __shfl_down(acc, off, 32);
    if (lane == 0)
        pgen[row] = 1.0f / (1.0f + __expf(-(acc + b_gen[0])));
}

// ---------------------------------------------------------------------------
extern "C" void kernel_launch(void* const* d_in, const int* in_sizes, int n_in,
                              void* d_out, int out_size, void* d_ws, size_t ws_size,
                              hipStream_t stream) {
    (void)in_sizes; (void)n_in; (void)out_size; (void)ws_size;

    const float* dec   = (const float*)d_in[0];   // (B,T,D)
    const float* enc   = (const float*)d_in[1];   // (B,S,D)
    const int*   mask  = (const int*)  d_in[2];   // (B,S)
    const float* w_ptr = (const float*)d_in[3];   // (2D,)
    const float* b_ptr = (const float*)d_in[4];   // (1,)
    const float* w_gen = (const float*)d_in[5];   // (2D,)
    const float* b_gen = (const float*)d_in[6];   // (1,)

    float* out  = (float*)d_out;
    float* pw   = out;                                    // (B,T,S)
    float* pgen = out + (size_t)Bc * Tc * Sc;             // (B,T,1)
    float* ctx  = pgen + (size_t)Bc * Tc;                 // (B,T,D)

    // workspace layout (16B-aligned sections)
    float*    enc_score = (float*)d_ws;                            // B*S f32
    float*    dec_score = enc_score + (size_t)Bc * Sc;             // B*T f32
    _Float16* pw16      = (_Float16*)(dec_score + (size_t)Bc * Tc);// B*T*S f16
    _Float16* e16t      = pw16 + (size_t)Bc * Tc * Sc;             // B*D*S f16

    // K1: GEMVs (40960 wave-rows, 8 waves/block)
    k1_scores<<<(Bc * Sc + Bc * Tc) / 8, 256, 0, stream>>>(dec, enc, w_ptr,
                                                           enc_score, dec_score);
    // K2b: encoder f32 -> f16 transpose ([b][d][s]); independent of K1/K2
    k2b_convT<<<Bc * (Sc / 64) * (Dc / 64), 256, 0, stream>>>(enc, e16t);
    // K2: masked softmax per (b,t) row; writes f32 output + f16 shadow
    k2_softmax<<<Bc * Tc, 256, 0, stream>>>(enc_score, dec_score, mask, b_ptr,
                                            pw, pw16);
    // K3: batched WMMA GEMM -> context (16 * 4 * 6 = 384 blocks)
    k3_ctx_gemm<<<Bc * 24, 256, 0, stream>>>(pw16, e16t, ctx);
    // K4: p_gen
    k4_pgen<<<(Bc * Tc) / 8, 256, 0, stream>>>(dec, ctx, w_gen, b_gen, pgen);
}
